// MutanFusion_23313082482846
// MI455X (gfx1250) — matrix operand order repaired and checked by reading, not compiled, not run on hardware
//
#include <hip/hip_runtime.h>

// Problem constants (from reference)
#define B_  512
#define D1_ 2048
#define D2_ 2400
#define H_  1200
#define R_  5

// Tiling
#define WG_THREADS 256          // 8 waves (wave32)
#define BM 128                  // batch rows per workgroup (16 per wave)
#define BN 16                   // H columns per workgroup
#define KT 32                   // K per WMMA step (bf16)
#define LDAF 36                 // padded f32 stride for A rows (144B, 16B aligned)
#define LDB  40                 // padded ushort stride for B columns (80B, 16B aligned)

typedef __bf16 bf16;
typedef __attribute__((ext_vector_type(16))) bf16  v16bf;
typedef __attribute__((ext_vector_type(8)))  float v8f;
typedef __attribute__((ext_vector_type(4)))  unsigned int v4u;

union BFrag {                    // 32 bytes: two 16B LDS loads -> one v16bf operand
    v4u   u[2];
    v16bf v;
};
union AFrag {                    // 16 f32 converted+packed -> one v16bf operand
    unsigned int u[8];
    v16bf v;
};

// Pack two f32 -> two bf16 (RNE) in one dword using the CDNA5 mixed-precision
// FMA converters (VOP3P opcodes 62/63): bf16(x) = fma_mix(x, 1.0, 0).
// 2 VALU ops per dword vs ~7 for the integer-rounding sequence.
__device__ __forceinline__ unsigned int pk2bf(float lo, float hi) {
    unsigned int d;
    asm("v_fma_mixlo_bf16 %0, %1, 1.0, 0" : "=v"(d) : "v"(lo));
    asm("v_fma_mixhi_bf16 %0, %1, 1.0, 0" : "+v"(d) : "v"(hi));
    return d;
}

// Single f32 -> bf16 (RNE) in the low 16 bits.
__device__ __forceinline__ unsigned short f2bf(float f) {
    unsigned int d;
    asm("v_fma_mixlo_bf16 %0, %1, 1.0, 0" : "=v"(d) : "v"(f));
    return (unsigned short)d;
}

__device__ __forceinline__ unsigned int lds_off(const void* p) {
    // LDS aperture keeps the byte offset in addr[31:0]
    return (unsigned int)(size_t)p;
}

// Issue async fill of one 128x32 f32 A tile into LDS (f32, padded rows).
// 1024 float4 slots, 4 per thread -> 4 global_load_async_to_lds_b128 each.
__device__ __forceinline__ void issueA_async(const float* __restrict__ X, int D,
                                             int bBase, int k0,
                                             float* ldsA, int tid)
{
    #pragma unroll
    for (int i = 0; i < 4; ++i) {
        int idx = tid + i * WG_THREADS;     // 0..1023
        int row = idx >> 3;                 // 128 rows
        int c4  = idx & 7;                  // 8 float4 per row
        const float* g = X + (size_t)(bBase + row) * D + k0 + c4 * 4;
        unsigned int l = lds_off(&ldsA[row * LDAF + c4 * 4]);
        asm volatile("global_load_async_to_lds_b128 %0, %1, off"
                     :: "v"(l), "v"(g) : "memory");
    }
}

// Synchronous W stage: 32x16 f32 -> bf16 LDS column-major [n][k] (tiny: 512 elems)
__device__ __forceinline__ void stageW(const float* __restrict__ W, int k0,
                                       int hBase, unsigned short* ldsB, int tid)
{
    const int wn = tid & 15;                // column n (h): coalesced across lanes
    const int wk = tid >> 4;                // k (0..15), also loads k+16
    float w0 = W[(size_t)(k0 + wk)      * H_ + hBase + wn];
    float w1 = W[(size_t)(k0 + wk + 16) * H_ + hBase + wn];
    ldsB[wn * LDB + wk]      = f2bf(w0);
    ldsB[wn * LDB + wk + 16] = f2bf(w1);
}

// One GEMM panel with async double buffering:
// acc(16x16 f32 per wave) += X[bBase:bBase+128, :] * W[:, hBase:hBase+16]
__device__ __forceinline__ v8f gemm_panel(const float* __restrict__ X,
                                          const float* __restrict__ W,
                                          int D,
                                          float* ldsA0, float* ldsA1,
                                          unsigned short* ldsB0, unsigned short* ldsB1,
                                          int bBase, int hBase, int tid)
{
    const int lane = tid & 31;
    const int wave = tid >> 5;
    const int m    = lane & 15;        // A row / B column within tile
    const int khi  = lane >> 4;        // 0 or 1: which K half this lane holds
    const int arow = wave * 16 + m;

    float*          bufA[2] = { ldsA0, ldsA1 };
    unsigned short* bufB[2] = { ldsB0, ldsB1 };

    v8f acc = {0.f, 0.f, 0.f, 0.f, 0.f, 0.f, 0.f, 0.f};

    const int nk = D / KT;

    // Prologue: fill buffer 0
    issueA_async(X, D, bBase, 0, bufA[0], tid);
    stageW(W, 0, hBase, bufB[0], tid);
    asm volatile("s_wait_asynccnt 0x0" ::: "memory");
    __syncthreads();

    for (int i = 0; i < nk; ++i) {
        const int cur = i & 1;

        // Prefetch next k tile into the other buffer (uniform branch: EXEC stays full)
        if (i + 1 < nk) {
            issueA_async(X, D, bBase, (i + 1) * KT, bufA[cur ^ 1], tid);
            stageW(W, (i + 1) * KT, hBase, bufB[cur ^ 1], tid);
        }

        // ---- compute on current buffers ----
        const float*          A  = bufA[cur];
        const unsigned short* Bm = bufB[cur];

        // A fragment: rows partitioned per wave; f32 from LDS, fma_mix-convert to bf16.
        // VGPR0-3 = K {0..7 | 8..15}, VGPR4-7 = K {16..23 | 24..31}
        const float4 fa0 = *reinterpret_cast<const float4*>(&A[arow * LDAF + khi * 8]);
        const float4 fa1 = *reinterpret_cast<const float4*>(&A[arow * LDAF + khi * 8 + 4]);
        const float4 fa2 = *reinterpret_cast<const float4*>(&A[arow * LDAF + 16 + khi * 8]);
        const float4 fa3 = *reinterpret_cast<const float4*>(&A[arow * LDAF + 16 + khi * 8 + 4]);
        AFrag a;
        a.u[0] = pk2bf(fa0.x, fa0.y);  a.u[1] = pk2bf(fa0.z, fa0.w);
        a.u[2] = pk2bf(fa1.x, fa1.y);  a.u[3] = pk2bf(fa1.z, fa1.w);
        a.u[4] = pk2bf(fa2.x, fa2.y);  a.u[5] = pk2bf(fa2.z, fa2.w);
        a.u[6] = pk2bf(fa3.x, fa3.y);  a.u[7] = pk2bf(fa3.z, fa3.w);

        // B fragment: lane n holds K {0..7 | 8..15} then K {16..23 | 24..31} (column-major bf16)
        BFrag b;
        b.u[0] = *reinterpret_cast<const v4u*>(&Bm[m * LDB + khi * 8]);
        b.u[1] = *reinterpret_cast<const v4u*>(&Bm[m * LDB + 16 + khi * 8]);

        acc = __builtin_amdgcn_wmma_f32_16x16x32_bf16(
                  /*neg_a=*/false, a.v, /*neg_b=*/false, b.v,
                  /*c_mod=*/(short)0, acc, /*reuse_a=*/false, /*reuse_b=*/false);

        // Drain this wave's async fills of the next buffer, then WG barrier:
        // after it, next buffer is visible to all and current buffer is free to refill.
        if (i + 1 < nk) {
            asm volatile("s_wait_asynccnt 0x0" ::: "memory");
        }
        __syncthreads();
    }
    return acc;
}

__global__ __launch_bounds__(WG_THREADS)
void mutan_fused_kernel(const float* __restrict__ in1, const float* __restrict__ in2,
                        const float* __restrict__ W1,  const float* __restrict__ b1,
                        const float* __restrict__ W2,  const float* __restrict__ b2,
                        float* __restrict__ out)
{
    __shared__ float          ldsA[2][BM * LDAF];   // 2 x 18 KB (f32, async-filled)
    __shared__ unsigned short ldsB[2][BN * LDB];    // 2 x 1.25 KB (bf16, col-major)

    const int tid   = threadIdx.x;
    const int lane  = tid & 31;
    const int wave  = tid >> 5;
    const int hBase = blockIdx.x * BN;
    const int bBase = blockIdx.y * BM;
    const int n     = lane & 15;                    // output column within tile

    v8f accO = {0.f, 0.f, 0.f, 0.f, 0.f, 0.f, 0.f, 0.f};

    for (int r = 0; r < R_; ++r) {
        v8f h1 = gemm_panel(in1, W1 + (size_t)r * D1_ * H_, D1_,
                            ldsA[0], ldsA[1], ldsB[0], ldsB[1], bBase, hBase, tid);
        v8f h2 = gemm_panel(in2, W2 + (size_t)r * D2_ * H_, D2_,
                            ldsA[0], ldsA[1], ldsB[0], ldsB[1], bBase, hBase, tid);
        // bias depends only on h (column N) -> per-lane scalar across all 8 acc regs
        const float bias1 = b1[r * H_ + hBase + n];
        const float bias2 = b2[r * H_ + hBase + n];
        #pragma unroll
        for (int i = 0; i < 8; ++i)
            accO[i] += (h1[i] + bias1) * (h2[i] + bias2);
    }

    // C/D layout: VGPR i -> row M = (lane>=16 ? 8 : 0) + i, col N = lane&15
    const int colg  = hBase + n;
    const int rbase = bBase + wave * 16 + (lane >> 4) * 8;
    #pragma unroll
    for (int i = 0; i < 8; ++i)
        out[(size_t)(rbase + i) * H_ + colg] = accO[i];
}

extern "C" void kernel_launch(void* const* d_in, const int* in_sizes, int n_in,
                              void* d_out, int out_size, void* d_ws, size_t ws_size,
                              hipStream_t stream) {
    const float* in1 = (const float*)d_in[0];
    const float* in2 = (const float*)d_in[1];
    const float* W1  = (const float*)d_in[2];
    const float* b1  = (const float*)d_in[3];
    const float* W2  = (const float*)d_in[4];
    const float* b2  = (const float*)d_in[5];
    float* out = (float*)d_out;

    dim3 grid(H_ / BN, B_ / BM);   // (75, 4) = 300 workgroups
    mutan_fused_kernel<<<grid, WG_THREADS, 0, stream>>>(in1, in2, W1, b1, W2, b2, out);
}
